// CliffordKANLayer_28381143892769
// MI455X (gfx1250) — matrix operand order repaired and checked by reading, compiled, not run on hardware
//
#include <hip/hip_runtime.h>
#include <hip/hip_bf16.h>

typedef __attribute__((ext_vector_type(16))) __bf16 v16bf;
typedef __attribute__((ext_vector_type(8)))  __bf16 v8bf;
typedef __attribute__((ext_vector_type(2)))  __bf16 v2bf;
typedef __attribute__((ext_vector_type(8)))  float  v8f;

#define B_DIM 64
#define I_DIM 64
#define O_DIM 64
#define G_DIM 4096
#define X_DIM 4
#define GB    256    // g-chunk per workgroup
#define KSTEP 32     // bf16 WMMA K per step
#define NSTEPS (GB / KSTEP)
#define A_STR 264    // A row stride in bf16 elems (528B = 16B*33, odd multiple -> conflict-free)
#define B_STR 40     // B row stride in bf16 elems (80B  = 16B*5,  odd multiple -> conflict-free)

// ---------------------------------------------------------------------------
// Kernel 1: init out[b,o,z] with the SiLU/Cayley branch + bias.
//   out[b,o,z] = sum_i ( sum_{x,y} sw[i,o,x]*silu(xin[b,i,y])*C[x,y,z] + sb[i,o,z] )
// Cayley table for Cl(2,0) hardcoded (basis 1, e1, e2, e12).
// ---------------------------------------------------------------------------
__global__ void __launch_bounds__(256)
ckan_silu_init(const float* __restrict__ xin,
               const float* __restrict__ sw,
               const float* __restrict__ sb,
               float* __restrict__ out) {
    int t = blockIdx.x * blockDim.x + threadIdx.x;   // 0 .. B*O-1
    if (t >= B_DIM * O_DIM) return;
    int b = t >> 6;
    int o = t & 63;
    float z0 = 0.f, z1 = 0.f, z2 = 0.f, z3 = 0.f;
    for (int i = 0; i < I_DIM; ++i) {
        const float* xp = xin + ((b * I_DIM + i) << 2);
        float x0 = xp[0], x1 = xp[1], x2 = xp[2], x3 = xp[3];
        float s0 = x0 / (1.f + __expf(-x0));
        float s1 = x1 / (1.f + __expf(-x1));
        float s2 = x2 / (1.f + __expf(-x2));
        float s3 = x3 / (1.f + __expf(-x3));
        const float* wp = sw + ((i * O_DIM + o) << 2);
        float a0 = wp[0], a1 = wp[1], a2 = wp[2], a3 = wp[3];
        const float* bp = sb + ((i * O_DIM + o) << 2);
        z0 += a0 * s0 + a1 * s1 + a2 * s2 - a3 * s3 + bp[0];
        z1 += a0 * s1 + a1 * s0 - a2 * s3 + a3 * s2 + bp[1];
        z2 += a0 * s2 + a1 * s3 + a2 * s0 - a3 * s1 + bp[2];
        z3 += a0 * s3 + a1 * s2 - a2 * s1 + a3 * s0 + bp[3];
    }
    float4 r; r.x = z0; r.y = z1; r.z = z2; r.w = z3;
    *(float4*)(out + (t << 2)) = r;
}

// ---------------------------------------------------------------------------
// Kernel 2: fused RBF + GEMM, accumulated into out via atomics.
//   Per workgroup: fixed i, g-chunk of 256.
//   A (64 x 256, bf16)  = exp(-||x[b,i,:] - grid[g,:]||^2)
//   B (256 x 256, bf16) = weights[i, o, g, x]  (K = g, N = o*4+x)
//   C += A x B  via v_wmma_f32_16x16x32_bf16, atomically added to out.
// B is staged in LDS fragment-major (n rows of K) so every lane's 16
// K-values are two contiguous 16B chunks -> ds_load_b128 pairs.
// ---------------------------------------------------------------------------
__global__ void __launch_bounds__(256)
ckan_rbf_gemm(const float* __restrict__ xin,
              const float* __restrict__ gridp,
              const float* __restrict__ W,
              float* __restrict__ out) {
    __shared__ __bf16 a_sm[B_DIM * A_STR];          // 33.0 KB
    __shared__ __bf16 b_sm[O_DIM * X_DIM * B_STR];  // 20.0 KB
    __shared__ float  xs[B_DIM * X_DIM];            //  1.0 KB

    const int t    = threadIdx.x;       // 0..255
    const int i    = blockIdx.y;        // input channel
    const int g0   = blockIdx.x * GB;   // g-chunk base
    const int lane = t & 31;
    const int wave = t >> 5;
    const int wm   = (wave & 1) * 32;   // wave M offset (2 m-tiles of 16)
    const int wn   = (wave >> 1) * 64;  // wave N offset (4 n-tiles of 16)

    // stage x[:, i, :] (256 floats) into LDS
    xs[t] = xin[(((t >> 2) * I_DIM + i) << 2) + (t & 3)];
    // this thread's grid point (it owns column g = g0 + t of the A tile)
    const float* gp = gridp + ((g0 + t) << 2);
    float gx = gp[0], gy = gp[1], gz = gp[2], gw = gp[3];
    __syncthreads();

    // Phase 1: RBF tile -> bf16 A operand in LDS (row-major, padded stride)
    #pragma unroll 4
    for (int b = 0; b < B_DIM; ++b) {
        float d0 = xs[(b << 2) + 0] - gx;
        float d1 = xs[(b << 2) + 1] - gy;
        float d2 = xs[(b << 2) + 2] - gz;
        float d3 = xs[(b << 2) + 3] - gw;
        float s  = d0 * d0 + d1 * d1 + d2 * d2 + d3 * d3;
        a_sm[b * A_STR + t] = (__bf16)__expf(-s);   // RHO = 1
    }

    v8f acc[2][4];
    #pragma unroll
    for (int mt = 0; mt < 2; ++mt)
        #pragma unroll
        for (int nt = 0; nt < 4; ++nt)
            acc[mt][nt] = (v8f){0.f, 0.f, 0.f, 0.f, 0.f, 0.f, 0.f, 0.f};

    const long ibase = (long)i * O_DIM * G_DIM * X_DIM;
    const int  colb  = wn + (lane & 15);   // this lane's N column (per n-tile +16)
    const int  krow  = (lane >> 4) << 3;   // K half: lanes 0-15 -> K 0..7,16..23; 16-31 -> 8..15,24..31
    const int  khalf = (lane >> 4) << 3;   // A-fragment K base

    for (int ks = 0; ks < NSTEPS; ++ks) {
        __syncthreads();   // a_sm ready (ks==0) / b_sm reads of prev step done

        // Stage B sub-tile: 32(k) x 256(n) fp32 -> bf16, fragment-major.
        // Unit u = (o, kpair): load 2 adjacent float4 (32B contiguous), emit
        // 4 packed bf16-pair stores (4B each), conflict-free under B_STR=40.
        #pragma unroll
        for (int it = 0; it < 4; ++it) {
            int u  = (it << 8) + t;      // 0..1023
            int o  = u >> 4;
            int kp = u & 15;
            const float* wp = W + ibase + (long)o * (G_DIM * X_DIM)
                              + ((long)(g0 + (ks << 5) + (kp << 1)) << 2);
            float4 f0 = *(const float4*)(wp);
            float4 f1 = *(const float4*)(wp + 4);
            __bf16* bb = b_sm + (o << 2) * B_STR + (kp << 1);
            v2bf p;
            p[0] = (__bf16)f0.x; p[1] = (__bf16)f1.x; *(v2bf*)(bb)             = p;
            p[0] = (__bf16)f0.y; p[1] = (__bf16)f1.y; *(v2bf*)(bb + B_STR)     = p;
            p[0] = (__bf16)f0.z; p[1] = (__bf16)f1.z; *(v2bf*)(bb + 2 * B_STR) = p;
            p[0] = (__bf16)f0.w; p[1] = (__bf16)f1.w; *(v2bf*)(bb + 3 * B_STR) = p;
        }
        // prefetch next sub-tile: 64 o-blocks x 512B; thread t grabs one 128B line
        if (ks + 1 < NSTEPS) {
            const float* p = W + ibase + (long)(t >> 2) * (G_DIM * X_DIM)
                             + (long)((g0 + ((ks + 1) << 5)) << 2) + ((t & 3) << 5);
            __builtin_prefetch(p, 0, 1);
        }
        __syncthreads();

        // B fragments: lane = column colb + nt*16, reads K krow..krow+7, krow+16..krow+23
        // as two aligned 16B chunks -> ds_load_b128 x2.
        v16bf bfrag[4];
        #pragma unroll
        for (int nt = 0; nt < 4; ++nt) {
            const __bf16* bp = b_sm + (colb + (nt << 4)) * B_STR + krow;
            v8bf lo = *(const v8bf*)(bp);
            v8bf hi = *(const v8bf*)(bp + 16);
            bfrag[nt] = __builtin_shufflevector(lo, hi,
                0, 1, 2, 3, 4, 5, 6, 7, 8, 9, 10, 11, 12, 13, 14, 15);
        }

        // A fragments (ISA 7.12.2 bf16 A 16x32): lane's 16 elems are
        // a_row[khalf..khalf+7] ++ a_row[16+khalf..16+khalf+7].
        #pragma unroll
        for (int mt = 0; mt < 2; ++mt) {
            const int row = wm + (mt << 4) + (lane & 15);
            const __bf16* ap = a_sm + row * A_STR + (ks << 5) + khalf;
            v8bf alo = *(const v8bf*)(ap);
            v8bf ahi = *(const v8bf*)(ap + 16);
            v16bf afrag = __builtin_shufflevector(alo, ahi,
                0, 1, 2, 3, 4, 5, 6, 7, 8, 9, 10, 11, 12, 13, 14, 15);
            #pragma unroll
            for (int nt = 0; nt < 4; ++nt)
                acc[mt][nt] = __builtin_amdgcn_wmma_f32_16x16x32_bf16(
                    false, afrag, false, bfrag[nt],
                    (short)0, acc[mt][nt], false, false);
        }
    }

    // C/D layout: VGPR r -> M = r (lanes 0-15) or M = 8+r (lanes 16-31); N = lane%16
    const int rowoff = ((lane >> 4) << 3);
    const int col    = lane & 15;
    #pragma unroll
    for (int mt = 0; mt < 2; ++mt) {
        int row0 = wm + (mt << 4) + rowoff;
        #pragma unroll
        for (int nt = 0; nt < 4; ++nt) {
            int n = wn + (nt << 4) + col;
            #pragma unroll
            for (int r = 0; r < 8; ++r)
                atomicAdd(&out[(row0 + r) * (O_DIM * X_DIM) + n], acc[mt][nt][r]);
        }
    }
}

// ---------------------------------------------------------------------------
extern "C" void kernel_launch(void* const* d_in, const int* in_sizes, int n_in,
                              void* d_out, int out_size, void* d_ws, size_t ws_size,
                              hipStream_t stream) {
    const float* x     = (const float*)d_in[0];  // (B, I, X)
    const float* gridp = (const float*)d_in[1];  // (G, X)
    const float* W     = (const float*)d_in[2];  // (I, O, G, X)
    const float* sw    = (const float*)d_in[3];  // (I, O, X)
    const float* sb    = (const float*)d_in[4];  // (I, O, X)
    float* out = (float*)d_out;                  // (B, O, X)

    ckan_silu_init<<<(B_DIM * O_DIM) / 256, 256, 0, stream>>>(x, sw, sb, out);

    dim3 g(G_DIM / GB, I_DIM);                   // 16 x 64 = 1024 workgroups
    ckan_rbf_gemm<<<g, 256, 0, stream>>>(x, gridp, W, out);
}
